// VGAEModel_79267916415284
// MI455X (gfx1250) — matrix operand order repaired and checked by reading, not compiled, not run on hardware
//
#include <hip/hip_runtime.h>
#include <math.h>

#define NN 16384
#define EE 524288

typedef __attribute__((ext_vector_type(16))) _Float16 v16h;
typedef __attribute__((ext_vector_type(8)))  _Float16 v8h;
typedef __attribute__((ext_vector_type(8)))  float    v8f;

// ---------------------------------------------------------------- utilities

__global__ void zero_f32(float* __restrict__ p, long n) {
  long i = (long)blockIdx.x * blockDim.x + threadIdx.x;
  if (i < n) p[i] = 0.f;
}

__global__ void degree_count(const int* __restrict__ src, const int* __restrict__ dst,
                             float* __restrict__ dout, float* __restrict__ din, int e) {
  int i = blockIdx.x * blockDim.x + threadIdx.x;
  if (i < e) {
    atomicAdd(&dout[src[i]], 1.f);
    atomicAdd(&din[dst[i]], 1.f);
  }
}

__global__ void norms_rsqrt(float* __restrict__ d, int n) {
  int i = blockIdx.x * blockDim.x + threadIdx.x;
  if (i < n) d[i] = rsqrtf(fmaxf(d[i], 1.f));
}

// out[i,j] = f16( x[i,j] * s[i] ) ; x is read exactly once -> NT load
__global__ void scale_to_f16(const float* __restrict__ x, const float* __restrict__ s,
                             _Float16* __restrict__ out, int rows, int cols) {
  long i = (long)blockIdx.x * blockDim.x + threadIdx.x;
  if (i < (long)rows * cols) {
    int r = (int)(i / cols);
    float v = __builtin_nontemporal_load(&x[i]);
    out[i] = (_Float16)(v * s[r]);
  }
}

// W[k,O] row-major -> Wt[o,K] row-major, f32 -> f16
__global__ void transpose_to_f16(const float* __restrict__ W, _Float16* __restrict__ Wt,
                                 int K, int O) {
  long i = (long)blockIdx.x * blockDim.x + threadIdx.x;
  if (i < (long)K * O) {
    int k = (int)(i / O), o = (int)(i % O);
    Wt[(long)o * K + k] = (_Float16)W[i];
  }
}

// ------------------------------------------------------- WMMA f16 fragments
// CDNA5 wave32 layouts (CDNA5 ISA §7.12.2).

__device__ __forceinline__ v16h load_a_frag(const _Float16* __restrict__ A, int lda,
                                            int row, int kb, int lane) {
  int m = lane & 15, hi = lane >> 4;
  const _Float16* p = A + (size_t)(row + m) * lda + kb + hi * 8;
  v8h lo = *(const v8h*)p;          // k = kb + hi*8 + [0..7]   -> elems 0..7
  v8h hh = *(const v8h*)(p + 16);   // k = kb + 16 + hi*8 + ... -> elems 8..15
  v16h r;
#pragma unroll
  for (int i = 0; i < 8; ++i) { r[i] = lo[i]; r[i + 8] = hh[i]; }
  return r;
}

// B supplied transposed (Bt row-major [Ncols, K]); lane n holds column n,
// k = kb + 8*laneHi + [0..7] (elems 0..7), kb + 16 + 8*laneHi + [0..7] (elems 8..15).
__device__ __forceinline__ v16h load_bt_frag(const _Float16* __restrict__ Bt, int ldb,
                                             int col, int kb, int lane) {
  int n = lane & 15, hi = lane >> 4;
  const _Float16* p = Bt + (size_t)(col + n) * ldb + kb + hi * 8;
  v8h a0 = *(const v8h*)p;
  v8h a1 = *(const v8h*)(p + 16);
  v16h r;
#pragma unroll
  for (int i = 0; i < 8; ++i) { r[i] = a0[i]; r[i + 8] = a1[i]; }
  return r;
}

// C = A[M,K](f16) * Bt[Nc,K](f16)^T, f32 accumulate. One wave per 16x16 tile.
__global__ void gemm_f16_wmma(const _Float16* __restrict__ A, const _Float16* __restrict__ Bt,
                              float* __restrict__ C, int M, int Nc, int K) {
  int lane = threadIdx.x & 31;
  int w = blockIdx.x * (blockDim.x >> 5) + (threadIdx.x >> 5);
  int tilesN = Nc >> 4;
  int nTiles = (M >> 4) * tilesN;
  if (w >= nTiles) return;                 // wave-uniform: EXEC stays all-ones
  int tm = (w / tilesN) << 4;
  int tn = (w % tilesN) << 4;
  v8f c = {};
  for (int kb = 0; kb < K; kb += 32) {
    v16h a = load_a_frag(A, K, tm, kb, lane);
    v16h b = load_bt_frag(Bt, K, tn, kb, lane);
    c = __builtin_amdgcn_wmma_f32_16x16x32_f16(false, a, false, b, (short)0, c, false, false);
  }
  int n = lane & 15, hi = lane >> 4;
#pragma unroll
  for (int r = 0; r < 8; ++r)
    C[(size_t)(tm + hi * 8 + r) * Nc + tn + n] = c[r];
}

// ------------------------------------------------------------- edge scatter

template <int F>
__global__ void edge_scatter(const int* __restrict__ src, const int* __restrict__ dst,
                             const float* __restrict__ H, float* __restrict__ AGG, int e) {
  const int CH = F / 4;
  long tid = (long)blockIdx.x * blockDim.x + threadIdx.x;
  if (tid >= (long)e * CH) return;
  int ed = (int)(tid / CH);
  int j  = (int)(tid % CH) * 4;
  int s = src[ed], d = dst[ed];
  const float4 v = *(const float4*)(H + (long)s * F + j);
  float* ap = AGG + (long)d * F + j;
  atomicAdd(ap + 0, v.x);
  atomicAdd(ap + 1, v.y);
  atomicAdd(ap + 2, v.z);
  atomicAdd(ap + 3, v.w);
}

// ------------------------------------------------------------- epilogue ops

// Hh = f16( relu(AGG * din_is[row] + b[col]) )
__global__ void finish_layer1(const float* __restrict__ AGG, const float* __restrict__ din_is,
                              const float* __restrict__ b, _Float16* __restrict__ Hh,
                              int rows, int cols) {
  long i = (long)blockIdx.x * blockDim.x + threadIdx.x;
  if (i < (long)rows * cols) {
    int r = (int)(i / cols), c = (int)(i % cols);
    float v = AGG[i] * din_is[r] + b[c];
    Hh[i] = (_Float16)fmaxf(v, 0.f);
  }
}

// z = (A2*din+b2) + noise * exp(A3*din+b3); write f32 (d_out tail) + f16 copy
__global__ void finish_z(const float* __restrict__ A2, const float* __restrict__ A3,
                         const float* __restrict__ din_is,
                         const float* __restrict__ b2, const float* __restrict__ b3,
                         const float* __restrict__ noise,
                         float* __restrict__ z_out, _Float16* __restrict__ z_h,
                         int rows, int cols) {
  long i = (long)blockIdx.x * blockDim.x + threadIdx.x;
  if (i < (long)rows * cols) {
    int r = (int)(i / cols), c = (int)(i % cols);
    float mean = A2[i] * din_is[r] + b2[c];
    float ls   = A3[i] * din_is[r] + b3[c];
    float z = mean + noise[i] * __expf(ls);
    z_out[i] = z;
    z_h[i] = (_Float16)z;
  }
}

// adj = sigmoid(Zh @ Zh^T), K=64. One wave per 16x64 strip (4 tiles):
// A fragments loaded once, reused for 4 WMMA pairs; NT stores keep the
// 1 GB streamed output from evicting the L2-resident 2 MB z matrix.
__global__ void decode_wmma(const _Float16* __restrict__ Zh, float* __restrict__ out, int n) {
  int lane = threadIdx.x & 31;
  long w = (long)blockIdx.x * (blockDim.x >> 5) + (threadIdx.x >> 5);
  int stripsN = n >> 6;                           // 4x 16-wide tiles per strip
  long nStrips = (long)(n >> 4) * stripsN;
  if (w >= nStrips) return;                       // wave-uniform
  int tm  = (int)(w / stripsN) << 4;
  int tn0 = (int)(w % stripsN) << 6;

  v16h a0 = load_a_frag(Zh, 64, tm, 0, lane);
  v16h a1 = load_a_frag(Zh, 64, tm, 32, lane);

  v8f c[4];
#pragma unroll
  for (int t = 0; t < 4; ++t) {
    v16h b0 = load_bt_frag(Zh, 64, tn0 + t * 16, 0, lane);
    v16h b1 = load_bt_frag(Zh, 64, tn0 + t * 16, 32, lane);
    v8f acc = {};
    acc = __builtin_amdgcn_wmma_f32_16x16x32_f16(false, a0, false, b0, (short)0, acc, false, false);
    acc = __builtin_amdgcn_wmma_f32_16x16x32_f16(false, a1, false, b1, (short)0, acc, false, false);
    c[t] = acc;
  }

  int nn = lane & 15, hi = lane >> 4;
#pragma unroll
  for (int r = 0; r < 8; ++r) {
    size_t rowoff = (size_t)(tm + hi * 8 + r) * n + nn;
#pragma unroll
    for (int t = 0; t < 4; ++t) {
      float v = c[t][r];
      __builtin_nontemporal_store(1.f / (1.f + __expf(-v)), &out[rowoff + tn0 + t * 16]);
    }
  }
}

// ---------------------------------------------------------------- dispatcher

extern "C" void kernel_launch(void* const* d_in, const int* in_sizes, int n_in,
                              void* d_out, int out_size, void* d_ws, size_t ws_size,
                              hipStream_t stream) {
  const float* features = (const float*)d_in[0];   // [N,512]
  const int*   src      = (const int*)d_in[1];     // [E]
  const int*   dst      = (const int*)d_in[2];     // [E]
  const float* noise    = (const float*)d_in[3];   // [N,64]
  const float* W1       = (const float*)d_in[4];   // [512,256]
  const float* b1       = (const float*)d_in[5];   // [256]
  const float* W2       = (const float*)d_in[6];   // [256,64]
  const float* b2       = (const float*)d_in[7];   // [64]
  const float* W3       = (const float*)d_in[8];   // [256,64]
  const float* b3       = (const float*)d_in[9];   // [64]

  float* adj_out = (float*)d_out;                          // [N,N]
  float* z_out   = (float*)d_out + (size_t)NN * NN;        // [N,64]

  // workspace carve-out (256B aligned)
  char* wp = (char*)d_ws;
  auto carve = [&](size_t bytes) -> void* {
    void* p = (void*)wp;
    wp += (bytes + 255) & ~(size_t)255;
    return p;
  };
  float*     deg   = (float*)carve((size_t)2 * NN * 4);        // dout | din
  float*     dout  = deg;
  float*     din   = deg + NN;
  _Float16*  Xs    = (_Float16*)carve((size_t)NN * 512 * 2);   // scaled feats f16
  _Float16*  W1t   = (_Float16*)carve((size_t)256 * 512 * 2);  // [256,512]
  _Float16*  W2t   = (_Float16*)carve((size_t)64 * 256 * 2);   // [64,256]
  _Float16*  W3t   = (_Float16*)carve((size_t)64 * 256 * 2);
  float*     H1    = (float*)carve((size_t)NN * 256 * 4);      // pre-agg layer1
  float*     AGG1  = (float*)carve((size_t)NN * 256 * 4);
  _Float16*  Hh    = (_Float16*)carve((size_t)NN * 256 * 2);   // post layer1 f16
  float*     H2    = (float*)carve((size_t)NN * 64 * 4);
  float*     H3    = (float*)carve((size_t)NN * 64 * 4);
  float*     AGG2  = (float*)carve((size_t)NN * 64 * 4);
  float*     AGG3  = (float*)carve((size_t)NN * 64 * 4);
  _Float16*  Zh    = (_Float16*)carve((size_t)NN * 64 * 2);

  const int T = 256;
  auto blocks = [&](long n) -> int { return (int)((n + T - 1) / T); };

  // 1) zero accumulators
  zero_f32<<<blocks((long)2 * NN), T, 0, stream>>>(deg, (long)2 * NN);
  zero_f32<<<blocks((long)NN * 256), T, 0, stream>>>(AGG1, (long)NN * 256);
  zero_f32<<<blocks((long)NN * 64), T, 0, stream>>>(AGG2, (long)NN * 64);
  zero_f32<<<blocks((long)NN * 64), T, 0, stream>>>(AGG3, (long)NN * 64);

  // 2) degree norms
  degree_count<<<blocks(EE), T, 0, stream>>>(src, dst, dout, din, EE);
  norms_rsqrt<<<blocks(2 * NN), T, 0, stream>>>(deg, 2 * NN);

  // 3) scale + convert activations / weights to f16
  scale_to_f16<<<blocks((long)NN * 512), T, 0, stream>>>(features, dout, Xs, NN, 512);
  transpose_to_f16<<<blocks((long)512 * 256), T, 0, stream>>>(W1, W1t, 512, 256);
  transpose_to_f16<<<blocks((long)256 * 64), T, 0, stream>>>(W2, W2t, 256, 64);
  transpose_to_f16<<<blocks((long)256 * 64), T, 0, stream>>>(W3, W3t, 256, 64);

  // 4) layer 1: GEMM -> edge scatter -> relu/bias/din, cvt f16
  {
    int nTiles = (NN / 16) * (256 / 16);            // 16384 waves
    gemm_f16_wmma<<<nTiles / 8, T, 0, stream>>>(Xs, W1t, H1, NN, 256, 512);
  }
  edge_scatter<256><<<blocks((long)EE * 64), T, 0, stream>>>(src, dst, H1, AGG1, EE);
  finish_layer1<<<blocks((long)NN * 256), T, 0, stream>>>(AGG1, din, b1, Hh, NN, 256);

  // 5) layers 2/3 (mean, log_std)
  {
    int nTiles = (NN / 16) * (64 / 16);             // 4096 waves
    gemm_f16_wmma<<<nTiles / 8, T, 0, stream>>>(Hh, W2t, H2, NN, 64, 256);
    gemm_f16_wmma<<<nTiles / 8, T, 0, stream>>>(Hh, W3t, H3, NN, 64, 256);
  }
  edge_scatter<64><<<blocks((long)EE * 16), T, 0, stream>>>(src, dst, H2, AGG2, EE);
  edge_scatter<64><<<blocks((long)EE * 16), T, 0, stream>>>(src, dst, H3, AGG3, EE);

  // 6) reparameterize: z (f32 tail of d_out) + f16 copy for decoder
  finish_z<<<blocks((long)NN * 64), T, 0, stream>>>(AGG2, AGG3, din, b2, b3, noise,
                                                    z_out, Zh, NN, 64);

  // 7) dense decode sigmoid(z z^T): 262144 strip-waves (16x64 each), 8/block
  {
    long nStrips = (long)(NN / 16) * (NN / 64);     // 262,144
    decode_wmma<<<(int)(nStrips / 8), T, 0, stream>>>(Zh, adj_out, NN);
  }
}
